// OuterProductMean_31361851195882
// MI455X (gfx1250) — matrix operand order repaired and checked by reading, not compiled, not run on hardware
//
#include <hip/hip_runtime.h>

#define NSEQ 128
#define NRES 256
#define CM   256
#define CC   32
#define CZ   128

typedef __bf16 bf16;
typedef unsigned int u32;
typedef __attribute__((ext_vector_type(16))) __bf16 v16bf;
typedef __attribute__((ext_vector_type(8)))  __bf16 v8bf;
typedef __attribute__((ext_vector_type(8)))  float  v8f;
typedef __attribute__((ext_vector_type(4)))  u32    v4u;
typedef __attribute__((ext_vector_type(8)))  int    v8i;
typedef __attribute__((ext_vector_type(4)))  int    v4i;

union V16 { v16bf v; v8bf h[2]; };

// ---------------------------------------------------------------------------
// TDM: stage a 2D tile (rows x 128 bf16, row-major, stride 128) from global
// into LDS at byte offset lds_off.  D# built per CDNA5 ISA 8.3/8.4.
// This toolchain exposes the 6-arg builtin (clang-23 / therock headers).
// ---------------------------------------------------------------------------
__device__ inline void tdm_load_2d_bf16(const bf16* gptr, u32 lds_off, u32 rows,
                                        u32 rowlen) {
    unsigned long long ga = (unsigned long long)(size_t)gptr;
    // Group 0: [1:0]=count=1, [63:32]=lds_addr, [120:64]=global_addr, [127:126]=type=2
    v4u g0 = { 1u,
               lds_off,
               (u32)(ga & 0xffffffffu),
               (u32)((ga >> 32) & 0x01ffffffu) | (2u << 30) };
    // Group 1:
    //  [17:16] data_size=1 (2 bytes)   [79:48] tensor_dim0 = rowlen
    //  [111:80] tensor_dim1 = rows     [127:112] tile_dim0 = rowlen
    //  [143:128] tile_dim1 = rows      [207:160] tensor_dim0_stride = rowlen
    v8i g1 = { (int)(1u << 16),
               (int)((rowlen & 0xffffu) << 16),
               (int)(((rowlen >> 16) & 0xffffu) | ((rows & 0xffffu) << 16)),
               (int)(((rows >> 16) & 0xffffu) | ((rowlen & 0xffffu) << 16)),
               (int)(rows & 0xffffu),
               (int)rowlen,
               0, 0 };
    v4i gz4 = { 0, 0, 0, 0 };
    v8i gz8 = { 0, 0, 0, 0, 0, 0, 0, 0 };
    __builtin_amdgcn_tensor_load_to_lds(g0, g1, gz4, gz4, gz8, 0);
}

// ---------------------------------------------------------------------------
// Kernel 0: Wout (1024 x 128 f32, ck x kz) -> WoutT (128 x 1024 bf16)
// ---------------------------------------------------------------------------
__global__ void woutT_kernel(const float* __restrict__ Wout, bf16* __restrict__ WoutT) {
    int idx = blockIdx.x * blockDim.x + threadIdx.x;           // kz*1024 + ck
    if (idx >= CZ * CC * CC) return;
    int kz = idx >> 10;
    int ck = idx & 1023;
    WoutT[idx] = (bf16)Wout[(size_t)ck * CZ + kz];
}

// ---------------------------------------------------------------------------
// Kernel 1: LayerNorm + both projections.  One wave32 per (s, i) row.
//   Amat[(i*32+c)][s] = a[s,i,c]   Bmat[(j*32+k)][s] = b[s,j,k]   (bf16)
// ---------------------------------------------------------------------------
__global__ __launch_bounds__(256)
void lnproj_kernel(const float* __restrict__ m, const float* __restrict__ gamma,
                   const float* __restrict__ beta, const float* __restrict__ W1,
                   const float* __restrict__ b1, const float* __restrict__ W2,
                   const float* __restrict__ b2,
                   bf16* __restrict__ Amat, bf16* __restrict__ Bmat) {
    const int wave = (blockIdx.x * blockDim.x + threadIdx.x) >> 5;
    const int lane = threadIdx.x & 31;
    const int s  = wave >> 8;
    const int ii = wave & 255;
    const float* row = m + (size_t)wave * CM;

    float x[8];
    float sum = 0.f, ss = 0.f;
#pragma unroll
    for (int e = 0; e < 8; ++e) {
        x[e] = row[lane * 8 + e];
        sum += x[e];
        ss  += x[e] * x[e];
    }
#pragma unroll
    for (int off = 16; off >= 1; off >>= 1) {
        sum += __shfl_xor(sum, off, 32);
        ss  += __shfl_xor(ss,  off, 32);
    }
    const float mu   = sum * (1.f / CM);
    const float var  = ss * (1.f / CM) - mu * mu;
    const float rinv = rsqrtf(var + 1e-5f);

    float mn[8];
#pragma unroll
    for (int e = 0; e < 8; ++e) {
        int d = lane * 8 + e;
        mn[e] = (x[e] - mu) * rinv * gamma[d] + beta[d];
    }

    for (int c = 0; c < CC; ++c) {
        float pa = 0.f, pb = 0.f;
#pragma unroll
        for (int e = 0; e < 8; ++e) {
            int d = lane * 8 + e;
            pa += mn[e] * W1[d * CC + c];
            pb += mn[e] * W2[d * CC + c];
        }
#pragma unroll
        for (int off = 16; off >= 1; off >>= 1) {
            pa += __shfl_xor(pa, off, 32);
            pb += __shfl_xor(pb, off, 32);
        }
        if (lane == 0) {
            Amat[(size_t)(ii * CC + c) * NSEQ + s] = (bf16)(pa + b1[c]);
            Bmat[(size_t)(ii * CC + c) * NSEQ + s] = (bf16)(pb + b2[c]);
        }
    }
}

// ---------------------------------------------------------------------------
// Kernel 2: fused outer-product GEMM (K = s) + output projection GEMM.
// LDS plan (128 KB total, reused):
//   stage 1: ldsA[256][128] bf16 (64 KB, offset 0) + ldsB[256][128] (64 KB, 65536)
//            both filled by one TDM tensor_load_to_lds each.
//   stage 2: same 128 KB re-laid as ldsO[64 pairs][1024] bf16.
// ---------------------------------------------------------------------------
__global__ __launch_bounds__(512)
void opm_kernel(const bf16* __restrict__ Amat, const bf16* __restrict__ Bmat,
                const bf16* __restrict__ WoutT, const float* __restrict__ bout,
                float* __restrict__ Z) {
    extern __shared__ char smem_raw[];
    bf16* ldsA = (bf16*)smem_raw;                 // [256][128]
    bf16* ldsB = ldsA + 256 * NSEQ;               // [256][128]
    bf16* ldsO = (bf16*)smem_raw;                 // [64][1024] (overlaps A+B)

    const int w     = threadIdx.x >> 5;           // wave id 0..15
    const int lane  = threadIdx.x & 31;
    const int lrow  = lane & 15;
    const int khalf = lane >> 4;
    const int bi = blockIdx.x;                    // i block (8 i's)
    const int bj = blockIdx.y;                    // j block (8 j's)

    // ---- TDM stage of A/B blocks into LDS (issued by wave 0 only) ----
    if (threadIdx.x == 0) {
        tdm_load_2d_bf16(Amat + (size_t)bi * (8 * CC) * NSEQ, 0u,     256u, 128u);
        tdm_load_2d_bf16(Bmat + (size_t)bj * (8 * CC) * NSEQ, 65536u, 256u, 128u);
        __builtin_amdgcn_s_wait_tensorcnt(0);
    }
    __syncthreads();

    // ---- Phase 1: O(256x256) = A_blk(256x128) x B_blk(128x256), K = s ----
    const v8f vzero = {0.f, 0.f, 0.f, 0.f, 0.f, 0.f, 0.f, 0.f};
    v8f acc[16];
#pragma unroll
    for (int t = 0; t < 16; ++t) acc[t] = vzero;

    const bf16* As = ldsA + (16 * w + lrow) * NSEQ;
#pragma unroll
    for (int kk = 0; kk < 4; ++kk) {
        const int k0 = kk * 32;
        V16 af;   // A 16x32 bf16 fragment: K contiguous per lane-half
        af.h[0] = *(const v8bf*)(As + k0 + khalf * 16);
        af.h[1] = *(const v8bf*)(As + k0 + khalf * 16 + 8);
#pragma unroll
        for (int t = 0; t < 16; ++t) {
            // B 32x16 fragment: col = lrow, K contiguous per lane-half
            v16bf bfrag = *(const v16bf*)(ldsB + (16 * t + lrow) * NSEQ
                                          + k0 + khalf * 16);
            acc[t] = __builtin_amdgcn_wmma_f32_16x16x32_bf16(
                false, af.v, false, bfrag, (short)0, acc[t], false, false);
        }
    }
    __syncthreads();   // all waves done reading ldsA/ldsB before overwrite

    // Round O tile to bf16 into LDS, relaid as [pair_local][c*32 + k].
    {
        const int li    = w >> 1;
        const int cbase = (w & 1) * 16;
        const int chalf = (lane < 16) ? 0 : 8;    // C/D layout: lanes 16-31 -> M+8
#pragma unroll
        for (int t = 0; t < 16; ++t) {
            const int lj = t >> 1;
            const int kb = (t & 1) * 16;
            bf16* dst = ldsO + (size_t)(li * 8 + lj) * 1024 + kb + lrow;
#pragma unroll
            for (int v = 0; v < 8; ++v) {
                dst[(size_t)(cbase + chalf + v) * 32] = (bf16)acc[t][v];
            }
        }
    }
    __syncthreads();

    // ---- Phase 2: Z(64 x 128) = ldsO(64x1024) x Wout(1024x128) ----
#pragma unroll
    for (int sub = 0; sub < 2; ++sub) {
        const int tile = w * 2 + sub;             // 0..31
        const int tr   = tile >> 3;               // pair-row tile 0..3
        const int tc   = tile & 7;                // kz-col tile 0..7
        v8f zacc = vzero;
        const bf16* aL = ldsO  + (size_t)(16 * tr + lrow) * 1024;
        const bf16* bG = WoutT + (size_t)(16 * tc + lrow) * 1024;
        for (int k0 = 0; k0 < 1024; k0 += 32) {
            V16 af;
            af.h[0] = *(const v8bf*)(aL + k0 + khalf * 16);
            af.h[1] = *(const v8bf*)(aL + k0 + khalf * 16 + 8);
            v16bf bfrag = *(const v16bf*)(bG + k0 + khalf * 16);
            zacc = __builtin_amdgcn_wmma_f32_16x16x32_bf16(
                false, af.v, false, bfrag, (short)0, zacc, false, false);
        }
        const int   kz = 16 * tc + lrow;
        const float bo = bout[kz] * (1.f / NSEQ);
        const int   prow0 = 16 * tr + ((lane < 16) ? 0 : 8);
#pragma unroll
        for (int v = 0; v < 8; ++v) {
            int pl = prow0 + v;                   // pair local 0..63
            int gi = bi * 8 + (pl >> 3);
            int gj = bj * 8 + (pl & 7);
            Z[((size_t)gi * NRES + gj) * CZ + kz] = zacc[v] * (1.f / NSEQ) + bo;
        }
    }
}

// ---------------------------------------------------------------------------
extern "C" void kernel_launch(void* const* d_in, const int* in_sizes, int n_in,
                              void* d_out, int out_size, void* d_ws, size_t ws_size,
                              hipStream_t stream) {
    (void)in_sizes; (void)n_in; (void)out_size; (void)ws_size;
    const float* m     = (const float*)d_in[0];
    const float* gamma = (const float*)d_in[1];
    const float* beta  = (const float*)d_in[2];
    const float* W1    = (const float*)d_in[3];
    const float* b1    = (const float*)d_in[4];
    const float* W2    = (const float*)d_in[5];
    const float* b2    = (const float*)d_in[6];
    const float* Wout  = (const float*)d_in[7];
    const float* bout  = (const float*)d_in[8];
    float* Z = (float*)d_out;

    char* ws = (char*)d_ws;
    bf16* Amat  = (bf16*)ws;                          // 8192 x 128 bf16 (2 MB)
    bf16* Bmat  = Amat + (size_t)8192 * NSEQ;         // 8192 x 128 bf16 (2 MB)
    bf16* WoutT = Bmat + (size_t)8192 * NSEQ;         // 128 x 1024 bf16 (256 KB)

    woutT_kernel<<<(CZ * CC * CC + 255) / 256, 256, 0, stream>>>(Wout, WoutT);
    lnproj_kernel<<<(NSEQ * NRES) / 8, 256, 0, stream>>>(m, gamma, beta, W1, b1,
                                                         W2, b2, Amat, Bmat);
    dim3 grid(NRES / 8, NRES / 8);
    opm_kernel<<<grid, 512, 64 * 1024 * sizeof(bf16), stream>>>(Amat, Bmat, WoutT,
                                                                bout, Z);
}
